// ExampleModel_58815282152186
// MI455X (gfx1250) — matrix operand order repaired and checked by reading, not compiled, not run on hardware
//
#include <hip/hip_runtime.h>

// Problem constants (match reference)
constexpr int Bb = 4, Tt = 1024, Dd = 1024, Hh = 2048, Ee = 8;
constexpr int Nn = Bb * Tt;            // 4096 tokens
constexpr int KCH = Dd / 32;           // 32 K-chunks of 32 for WMMA f16
constexpr int SWZ_ELEMS = KCH * 512;   // swizzled [D,16] weight matrix, 16384 halves

// Workspace byte offsets
constexpr size_t WS_W2S = 0;                                 // E*H floats (64 KB)
constexpr size_t WS_V   = WS_W2S + (size_t)Ee * Hh * 4;      // E*D floats (32 KB)
constexpr size_t WS_C   = WS_V   + (size_t)Ee * Dd * 4;      // E floats (pad 64 B)
constexpr size_t WS_SWZ = WS_C   + 64;                       // 16384 f16 (32 KB)
constexpr size_t WS_S   = WS_SWZ + (size_t)SWZ_ELEMS * 2;    // N floats (16 KB)

typedef __attribute__((ext_vector_type(16))) _Float16 v16h;
typedef __attribute__((ext_vector_type(8)))  float    v8f;
typedef __attribute__((ext_vector_type(4)))  float    v4f;   // native vector for NT loads

__device__ __forceinline__ float wave_reduce_sum(float v) {
#pragma unroll
    for (int off = 16; off > 0; off >>= 1) v += __shfl_xor(v, off, 32);
    return v;
}

__device__ __forceinline__ float blk_reduce_sum(float v, float* sm) {
    const int t = threadIdx.x;
    sm[t] = v; __syncthreads();
    for (int s = 128; s > 0; s >>= 1) {
        if (t < s) sm[t] += sm[t + s];
        __syncthreads();
    }
    float r = sm[0]; __syncthreads();
    return r;
}

// ---- w2s[e,h] = sum_d w2[e,h,d] : one wave per (e,h) row, b128 NT streaming ----
__global__ __launch_bounds__(256) void w2sum_kernel(const float* __restrict__ w2,
                                                    float* __restrict__ w2s) {
    const int wave = threadIdx.x >> 5, lane = threadIdx.x & 31;
    const int row  = blockIdx.x * 8 + wave;                 // row = e*H + h
    const v4f* p = (const v4f*)(w2 + (size_t)row * Dd) + lane;
    float v = 0.f;
#pragma unroll
    for (int i = 0; i < Dd / 128; ++i) {                    // 8 x b128 per lane
        const v4f a = __builtin_nontemporal_load(p + i * 32);
        v += (a.x + a.y) + (a.z + a.w);
    }
    v = wave_reduce_sum(v);
    if (lane == 0) w2s[row] = v;
}

// ---- v[e,d] = sum_h w1[e,d,h] * w2s[e,h] : one wave per (e,d) row ----
__global__ __launch_bounds__(256) void vmat_kernel(const float* __restrict__ w1,
                                                   const float* __restrict__ w2s,
                                                   float* __restrict__ vout) {
    const int wave = threadIdx.x >> 5, lane = threadIdx.x & 31;
    const int row  = blockIdx.x * 8 + wave;                 // row = e*D + d
    const int e    = row / Dd;
    const v4f* wp = (const v4f*)(w1 + (size_t)row * Hh) + lane;   // NT stream (single use)
    const v4f* sp = (const v4f*)(w2s + (size_t)e * Hh) + lane;    // hot in L2
    float v = 0.f;
#pragma unroll
    for (int i = 0; i < Hh / 128; ++i) {                    // 16 x b128 per lane
        const v4f a = __builtin_nontemporal_load(wp + i * 32);
        const v4f b = sp[i * 32];
        v = fmaf(a.x, b.x, v); v = fmaf(a.y, b.y, v);
        v = fmaf(a.z, b.z, v); v = fmaf(a.w, b.w, v);
    }
    v = wave_reduce_sum(v);
    if (lane == 0) vout[row] = v;
}

// ---- c[e] = b1[e]·w2s[e] + sum_d b2[e,d] ----
__global__ __launch_bounds__(256) void cvec_kernel(const float* __restrict__ b1,
                                                   const float* __restrict__ b2,
                                                   const float* __restrict__ w2s,
                                                   float* __restrict__ cvec) {
    __shared__ float sm[256];
    const int e = blockIdx.x;
    float v = 0.f;
    for (int i = threadIdx.x; i < Hh; i += 256)
        v = fmaf(b1[e * Hh + i], w2s[e * Hh + i], v);
    for (int i = threadIdx.x; i < Dd; i += 256) v += b2[e * Dd + i];
    v = blk_reduce_sum(v, sm);
    if (threadIdx.x == 0) cvec[e] = v;
}

// ---- Build the WMMA-B-fragment-swizzled f16 copy of [wg | v^T] (D x 16) ----
// B layout (16x16x32 f16): lanes 0-15 hold N=lane, K=k0+0..7,16..23; lanes 16-31 hold K=k0+8..15,24..31.
// swz[chunk*512 + lane*16 + j] = W[d = chunk*32 + (j<8 ? 8h+j : 16+8h+j-8), h=lane>>4][col = lane&15]
__global__ __launch_bounds__(256) void build_swz_kernel(const float* __restrict__ wg,
                                                        const float* __restrict__ v,
                                                        _Float16* __restrict__ swz) {
    const int idx = blockIdx.x * 256 + threadIdx.x;
    if (idx >= SWZ_ELEMS) return;
    const int chunk = idx >> 9;
    const int r     = idx & 511;
    const int l     = r >> 4;
    const int j     = r & 15;
    const int col   = l & 15;
    const int krel  = (j < 8) ? (8 * (l >> 4) + j) : (16 + 8 * (l >> 4) + (j - 8));
    const int d     = chunk * 32 + krel;
    const float val = (col < 8) ? wg[d * Ee + col] : v[(col - 8) * Dd + d];
    swz[idx] = (_Float16)val;
}

// ---- Token pass: P[N,16] = X @ [wg | v^T] via v_wmma_f32_16x16x32_f16,
//      then per-token softmax/top-2 gating and s[n]. 8 waves/block, 16 tokens/wave. ----
__global__ __launch_bounds__(256) void moe_score_kernel(const float* __restrict__ x,
                                                        const _Float16* __restrict__ swz,
                                                        const float* __restrict__ cvec,
                                                        float* __restrict__ sout) {
    __shared__ __align__(32) _Float16 sW[SWZ_ELEMS];  // 32 KB swizzled weights
    __shared__ float sP[8 * 256];                     // 8 waves x 16x16 f32 tile

    // cooperative 128b load of the swizzled weight matrix into LDS (reused by all waves)
    {
        const int4* src = (const int4*)swz;
        int4* dst = (int4*)sW;
        for (int i = threadIdx.x; i < SWZ_ELEMS / 8; i += 256) dst[i] = src[i];
    }
    __syncthreads();

    const int wave = threadIdx.x >> 5;
    const int lane = threadIdx.x & 31;
    const int half = lane >> 4;      // lane group (K split)
    const int col  = lane & 15;      // A row / B column / C column index
    const int m0   = blockIdx.x * 128 + wave * 16;   // first token of this wave's tile
    const float* xr = x + (size_t)(m0 + col) * Dd;   // A-row for this lane
    const int s0 = half ? 8 : 0;     // A layout: lane<16 -> K 0-7,16-23 ; lane>=16 -> K 8-15,24-31

    v8f acc = {};
    for (int chunk = 0; chunk < KCH; ++chunk) {
        const float* xp = xr + chunk * 32 + s0;
        const v4f a0 = *(const v4f*)(xp);
        const v4f a1 = *(const v4f*)(xp + 4);
        const v4f a2 = *(const v4f*)(xp + 16);
        const v4f a3 = *(const v4f*)(xp + 20);
        v16h av;
        av[0]  = (_Float16)a0.x; av[1]  = (_Float16)a0.y; av[2]  = (_Float16)a0.z; av[3]  = (_Float16)a0.w;
        av[4]  = (_Float16)a1.x; av[5]  = (_Float16)a1.y; av[6]  = (_Float16)a1.z; av[7]  = (_Float16)a1.w;
        av[8]  = (_Float16)a2.x; av[9]  = (_Float16)a2.y; av[10] = (_Float16)a2.z; av[11] = (_Float16)a2.w;
        av[12] = (_Float16)a3.x; av[13] = (_Float16)a3.y; av[14] = (_Float16)a3.z; av[15] = (_Float16)a3.w;
        const v16h bv = *(const v16h*)(sW + chunk * 512 + lane * 16);
        // D = A(16x32 f16) * B(32x16 f16) + C(f32)
        acc = __builtin_amdgcn_wmma_f32_16x16x32_f16(false, av, false, bv,
                                                     (short)0, acc, false, false);
    }

    // C layout: VGPR j holds row M = j + 8*half, col N = lane&15
    float* pw = sP + wave * 256;
#pragma unroll
    for (int j = 0; j < 8; ++j) pw[(j + half * 8) * 16 + col] = acc[j];
    __syncthreads();

    // Gating: lanes 0..15 each own one token of this wave's tile
    if (lane < 16) {
        const float* p = sP + wave * 256 + lane * 16;   // 8 gate logits + 8 expert sums
        float l0 = p[0]; int e0 = 0;
#pragma unroll
        for (int e = 1; e < Ee; ++e) if (p[e] > l0) { l0 = p[e]; e0 = e; }
        float l1 = -3.4e38f; int e1 = 0;
#pragma unroll
        for (int e = 0; e < Ee; ++e) if (e != e0 && p[e] > l1) { l1 = p[e]; e1 = e; }
        // normalized top-2 softmax gates (global softmax denom cancels)
        const float g0 = 1.f / (1.f + expf(l1 - l0));
        const float g1 = 1.f - g0;
        const float sv = g0 * (p[8 + e0] + cvec[e0]) + g1 * (p[8 + e1] + cvec[e1]);
        sout[m0 + lane] = sv;
    }
}

// ---- log_softmax over T per batch row ----
__global__ __launch_bounds__(256) void logsoftmax_kernel(const float* __restrict__ s,
                                                         float* __restrict__ out) {
    __shared__ float sm[256];
    const int b = blockIdx.x, t = threadIdx.x;
    const float* row = s + (size_t)b * Tt;
    float vals[4];
    float mx = -3.4e38f;
#pragma unroll
    for (int i = 0; i < 4; ++i) { vals[i] = row[t + i * 256]; mx = fmaxf(mx, vals[i]); }
    sm[t] = mx; __syncthreads();
    for (int r = 128; r > 0; r >>= 1) { if (t < r) sm[t] = fmaxf(sm[t], sm[t + r]); __syncthreads(); }
    const float m = sm[0]; __syncthreads();
    float lsum = 0.f;
#pragma unroll
    for (int i = 0; i < 4; ++i) lsum += expf(vals[i] - m);
    sm[t] = lsum; __syncthreads();
    for (int r = 128; r > 0; r >>= 1) { if (t < r) sm[t] += sm[t + r]; __syncthreads(); }
    const float lz = logf(sm[0]);
#pragma unroll
    for (int i = 0; i < 4; ++i) out[(size_t)b * Tt + t + i * 256] = vals[i] - m - lz;
}

extern "C" void kernel_launch(void* const* d_in, const int* in_sizes, int n_in,
                              void* d_out, int out_size, void* d_ws, size_t ws_size,
                              hipStream_t stream) {
    const float* x  = (const float*)d_in[0];
    const float* wg = (const float*)d_in[1];
    const float* w1 = (const float*)d_in[2];
    const float* b1 = (const float*)d_in[3];
    const float* w2 = (const float*)d_in[4];
    const float* b2 = (const float*)d_in[5];

    char* ws = (char*)d_ws;
    float*    w2s  = (float*)(ws + WS_W2S);
    float*    vbuf = (float*)(ws + WS_V);
    float*    cv   = (float*)(ws + WS_C);
    _Float16* swz  = (_Float16*)(ws + WS_SWZ);
    float*    sbuf = (float*)(ws + WS_S);
    float*    out  = (float*)d_out;

    w2sum_kernel   <<<Ee * Hh / 8, 256, 0, stream>>>(w2, w2s);
    vmat_kernel    <<<Ee * Dd / 8, 256, 0, stream>>>(w1, w2s, vbuf);
    cvec_kernel    <<<Ee,          256, 0, stream>>>(b1, b2, w2s, cv);
    build_swz_kernel<<<(SWZ_ELEMS + 255) / 256, 256, 0, stream>>>(wg, vbuf, swz);
    moe_score_kernel<<<Nn / 128, 256, 0, stream>>>(x, swz, cv, sbuf);
    logsoftmax_kernel<<<Bb, 256, 0, stream>>>(sbuf, out);
}
